// AverageHeadAttention_9689446219906
// MI455X (gfx1250) — compile-verified, matrix operand order checked by reading, not run on hardware
//
#include <hip/hip_runtime.h>
#include <hip/hip_bf16.h>
#include <math.h>

typedef __attribute__((ext_vector_type(16))) _Float16 v16h;
typedef __attribute__((ext_vector_type(8)))  _Float16 v8h;
typedef __attribute__((ext_vector_type(4)))  _Float16 v4h;
typedef __attribute__((ext_vector_type(8)))  float    v8f;
typedef __attribute__((ext_vector_type(4)))  float    v4f;

#define B_  2
#define L_  2048
#define D_  1024
#define H_  16
#define DK_ 64
#define DV_ 64

#define SROW 2052                       // padded f32 row stride of LDS score strip
#define SMEM_ATTN ((16 * SROW + 16 * 64) * 4)   // 135,424 B (< 320 KB/WGP)

// ---------------------------------------------------------------------------
// WMMA helpers (gfx1250 wave32, v_wmma_f32_16x16x32_f16)
// A operand: 16x32 (MxK) f16, lane = M%16; per-lane k = (j<8?0:16)+8*(lane/16)+j%8
// B operand: 32x16 (KxN) f16, same register pattern with lane = N  -> tile [n][k]
// C/D: 8 VGPRs f32, M = r + 8*(lane/16), N = lane%16
// ---------------------------------------------------------------------------
__device__ __forceinline__ v16h fill16(v8h lo, v8h hi) {
  v16h r;
#pragma unroll
  for (int i = 0; i < 8; ++i) { r[i] = lo[i]; r[8 + i] = hi[i]; }
  return r;
}

// Load one 16x32 operand from a row-major f16 tile ([m][k] for A, [n][k] for B).
// Requires stride*2 % 16 == 0 so b128 loads stay aligned.
__device__ __forceinline__ v16h load_op_f16(const _Float16* tile, int stride, int lane) {
  const int m  = lane & 15;
  const int hb = (lane >> 4) << 3;            // 0 or 8
  const _Float16* p = tile + m * stride + hb;
  v8h lo = *(const v8h*)(p);
  v8h hi = *(const v8h*)(p + 16);
  return fill16(lo, hi);
}

// Same, but source tile is f32 (converted on the fly) — used for softmaxed P.
__device__ __forceinline__ v16h load_op_f32(const float* tile, int stride, int lane) {
  const int m  = lane & 15;
  const int hb = (lane >> 4) << 3;
  const float* p = tile + m * stride + hb;
  v4f a0 = *(const v4f*)(p);
  v4f a1 = *(const v4f*)(p + 4);
  v4f b0 = *(const v4f*)(p + 16);
  v4f b1 = *(const v4f*)(p + 20);
  v16h r;
  r[0]  = (_Float16)a0.x; r[1]  = (_Float16)a0.y; r[2]  = (_Float16)a0.z; r[3]  = (_Float16)a0.w;
  r[4]  = (_Float16)a1.x; r[5]  = (_Float16)a1.y; r[6]  = (_Float16)a1.z; r[7]  = (_Float16)a1.w;
  r[8]  = (_Float16)b0.x; r[9]  = (_Float16)b0.y; r[10] = (_Float16)b0.z; r[11] = (_Float16)b0.w;
  r[12] = (_Float16)b1.x; r[13] = (_Float16)b1.y; r[14] = (_Float16)b1.z; r[15] = (_Float16)b1.w;
  return r;
}

__device__ __forceinline__ v8f wmma_f16(v16h a, v16h b, v8f c) {
  return __builtin_amdgcn_wmma_f32_16x16x32_f16(false, a, false, b, (short)0, c, false, false);
}

// ---------------------------------------------------------------------------
// Kernel 1: QKV projection.  out_h[b,h,l,d] (f16) = X[b,l,:] @ W[:, h*64+d] + bias
// X: [B*L, 1024] f32, W: [1024, 1024] f32.  Block 256 thr (8 waves),
// tile 128(M) x 64(N), K-steps of 32.  Grid (32, 16).
// ---------------------------------------------------------------------------
__global__ __launch_bounds__(256) void k_qkv(const float* __restrict__ X,
                                             const float* __restrict__ W,
                                             const float* __restrict__ bias,
                                             _Float16* __restrict__ outh) {
  __shared__ _Float16 At[128 * 40];   // [m][k], stride 40 (80B rows -> 16B aligned)
  __shared__ _Float16 Bt[64 * 40];    // [n][k]
  const int tid = threadIdx.x, lane = tid & 31, w = tid >> 5;
  const int m0 = blockIdx.x * 128, n0 = blockIdx.y * 64;
  v8f acc[4] = {};
  for (int kt = 0; kt < 32; ++kt) {
    __syncthreads();
    for (int i = tid; i < 1024; i += 256) {            // A: 128x32 f32 -> f16
      int r = i >> 3, c4 = (i & 7) << 2;
      v4f f = *(const v4f*)(X + (size_t)(m0 + r) * D_ + kt * 32 + c4);
      v4h hv; hv[0] = (_Float16)f.x; hv[1] = (_Float16)f.y;
              hv[2] = (_Float16)f.z; hv[3] = (_Float16)f.w;
      *(v4h*)(At + r * 40 + c4) = hv;
    }
    for (int i = tid; i < 2048; i += 256) {            // B^T: W[k][n] -> Bt[n][k]
      int k = i >> 6, n = i & 63;
      Bt[n * 40 + k] = (_Float16)W[(size_t)(kt * 32 + k) * 1024 + n0 + n];
    }
    __syncthreads();
    v16h a = load_op_f16(At + w * 16 * 40, 40, lane);
#pragma unroll
    for (int nt = 0; nt < 4; ++nt) {
      v16h bm = load_op_f16(Bt + nt * 16 * 40, 40, lane);
      acc[nt] = wmma_f16(a, bm, acc[nt]);
    }
  }
  const int mloc = (lane >> 4) << 3, nloc = lane & 15;
#pragma unroll
  for (int nt = 0; nt < 4; ++nt) {
    int col = n0 + nt * 16 + nloc;
    int h = col >> 6, d = col & 63;
    float bv = bias[col];
#pragma unroll
    for (int r = 0; r < 8; ++r) {
      int row = m0 + w * 16 + mloc + r;               // [0, B*L)
      int b = row >> 11, l = row & 2047;
      outh[(((size_t)b * H_ + h) * L_ + l) * 64 + d] = (_Float16)(acc[nt][r] + bv);
    }
  }
}

// ---------------------------------------------------------------------------
// Kernel 2: transpose V heads: vh [BH, L, 64] f16 -> vt [BH, 64, L] f16,
// so PV B-operands become direct global b128 loads.  Grid (32, 32), block 256.
// ---------------------------------------------------------------------------
__global__ __launch_bounds__(256) void k_vtrans(const _Float16* __restrict__ vh,
                                                _Float16* __restrict__ vt) {
  __shared__ _Float16 T[64 * 68];
  const int tid = threadIdx.x;
  const int bh = blockIdx.y;
  const int l0 = blockIdx.x * 64;
  const _Float16* src = vh + ((size_t)bh * L_ + l0) * 64;
  for (int i = tid; i < 1024; i += 256) {
    int l = i >> 4, d4 = (i & 15) << 2;
    v4h x = *(const v4h*)(src + l * 64 + d4);
    T[(d4 + 0) * 68 + l] = x[0];
    T[(d4 + 1) * 68 + l] = x[1];
    T[(d4 + 2) * 68 + l] = x[2];
    T[(d4 + 3) * 68 + l] = x[3];
  }
  __syncthreads();
  _Float16* dst = vt + (size_t)bh * 64 * L_ + l0;
  for (int i = tid; i < 1024; i += 256) {
    int d = i >> 4, l4 = (i & 15) << 2;
    *(v4h*)(dst + (size_t)d * L_ + l4) = *(const v4h*)(&T[d * 68 + l4]);
  }
}

// ---------------------------------------------------------------------------
// Kernel 3 (fused): scores -> softmax -> PV for a strip of 16 query rows of
// one (b,h).  Score strip [16][2048] f32 lives in LDS (CDNA5: 320KB/WGP).
//   phase 1: S = QK^T/8 (+mask) via WMMA, waves split the 2048 columns
//   phase 2: online softmax in LDS (v4f sweeps); P written to global once,
//            with non-temporal stores (P is never re-read from global)
//   phase 3: O = P@V via WMMA (A from LDS f32, B from transposed V in global),
//            waves split K; reduce via ds_add_f32 then weighted global atomics
// Grid (128, 32), block 256, dynamic LDS = SMEM_ATTN.
// ---------------------------------------------------------------------------
__global__ __launch_bounds__(256) void k_attn(const _Float16* __restrict__ qh,
                                              const _Float16* __restrict__ kh,
                                              const _Float16* __restrict__ vt,
                                              const unsigned char* __restrict__ mask,
                                              const float* __restrict__ aw_,
                                              float* __restrict__ attn,
                                              float* __restrict__ ctx) {
  extern __shared__ float smem[];
  float* S    = smem;                 // [16][SROW]
  float* Obuf = smem + 16 * SROW;     // [16][64]
  const int tid = threadIdx.x, lane = tid & 31, w = tid >> 5;
  const int bh = blockIdx.y, b = bh / H_, h = bh % H_;
  const int m0 = blockIdx.x * 16;
  const int mloc = (lane >> 4) << 3, nloc = lane & 15;

  for (int i = tid; i < 16 * 64; i += 256) Obuf[i] = 0.f;

  // ---- phase 1: scores ----
  const _Float16* qbase = qh + (((size_t)b * H_ + h) * L_ + m0) * 64;
  v16h a0 = load_op_f16(qbase, 64, lane);
  v16h a1 = load_op_f16(qbase + 32, 64, lane);
  const unsigned char* mbase = mask + ((size_t)b * L_ + m0) * L_;
  for (int nt = 0; nt < 16; ++nt) {
    int c0 = w * 256 + nt * 16;
    const _Float16* kbase = kh + (((size_t)b * H_ + h) * L_ + c0) * 64;
    v8f acc = {};
    acc = wmma_f16(a0, load_op_f16(kbase, 64, lane), acc);
    acc = wmma_f16(a1, load_op_f16(kbase + 32, 64, lane), acc);
    int c = c0 + nloc;
#pragma unroll
    for (int r = 0; r < 8; ++r) {
      int m = mloc + r;
      float s = acc[r] * 0.125f;                      // 1/sqrt(64)
      if (mbase[(size_t)m * L_ + c]) s = -__builtin_inff();
      S[m * SROW + c] = s;
    }
  }
  __syncthreads();

  // ---- phase 2: online softmax (wave w owns rows 2w, 2w+1) ----
  float* grow0 = attn + (((size_t)h * B_ + b) * L_ + m0) * L_;
  for (int rr = 0; rr < 2; ++rr) {
    int m = w * 2 + rr;
    float* srow = S + m * SROW;
    float mx = -__builtin_inff(), sum = 0.f;
    for (int i = lane * 4; i < L_; i += 128) {
      v4f x = *(const v4f*)(srow + i);
      float t  = fmaxf(fmaxf(x.x, x.y), fmaxf(x.z, x.w));
      float nm = fmaxf(mx, t);
      sum = sum * __expf(mx - nm) + __expf(x.x - nm) + __expf(x.y - nm)
                                  + __expf(x.z - nm) + __expf(x.w - nm);
      mx = nm;
    }
#pragma unroll
    for (int off = 16; off > 0; off >>= 1) {
      float m2 = __shfl_xor(mx, off, 32);
      float s2 = __shfl_xor(sum, off, 32);
      float nm = fmaxf(mx, m2);
      sum = sum * __expf(mx - nm) + s2 * __expf(m2 - nm);
      mx = nm;
    }
    float inv = 1.f / sum;
    float* gr = grow0 + (size_t)m * L_;
    for (int i = lane * 4; i < L_; i += 128) {
      v4f x = *(const v4f*)(srow + i);
      v4f p;
      p.x = __expf(x.x - mx) * inv; p.y = __expf(x.y - mx) * inv;
      p.z = __expf(x.z - mx) * inv; p.w = __expf(x.w - mx) * inv;
      *(v4f*)(srow + i) = p;                               // keep P in LDS for PV
      __builtin_nontemporal_store(p, (v4f*)(gr + i));      // attn output (once, NT)
    }
  }
  __syncthreads();

  // ---- phase 3: O += P@V over this wave's K range ----
  const _Float16* vbase = vt + ((size_t)b * H_ + h) * 64 * L_;
  v8f oacc[4] = {};
  for (int kt = 0; kt < 8; ++kt) {
    int k0 = w * 256 + kt * 32;
    v16h a = load_op_f32(S + k0, SROW, lane);
#pragma unroll
    for (int nt = 0; nt < 4; ++nt) {
      v16h bm = load_op_f16(vbase + (size_t)(nt * 16) * L_ + k0, L_, lane);
      oacc[nt] = wmma_f16(a, bm, oacc[nt]);
    }
  }
#pragma unroll
  for (int nt = 0; nt < 4; ++nt)
#pragma unroll
    for (int r = 0; r < 8; ++r)
      atomicAdd(&Obuf[(mloc + r) * 64 + nt * 16 + nloc], oacc[nt][r]);
  __syncthreads();

  // ---- phase 4: weighted accumulate into ctx ----
  const float aw = aw_[b * H_ + h];
  for (int i = tid; i < 16 * 64; i += 256) {
    int m = i >> 6, d = i & 63;
    atomicAdd(ctx + ((size_t)b * L_ + m0 + m) * 64 + d, aw * Obuf[i]);
  }
}

// ---------------------------------------------------------------------------
// Kernel 4: x = ctx @ Wfc + bfc + residual.  M=4096, N=1024, K=64.
// Tile 128x64, single K pass (2 WMMA steps).  Grid (32, 16), block 256.
// ---------------------------------------------------------------------------
__global__ __launch_bounds__(256) void k_fc(const float* __restrict__ ctx,
                                            const float* __restrict__ Wfc,
                                            const float* __restrict__ bfc,
                                            const float* __restrict__ resid,
                                            float* __restrict__ out) {
  __shared__ _Float16 At[128 * 80];                   // [m][k0..63], stride 80 (160B)
  __shared__ _Float16 Bt[64 * 80];                    // [n][k0..63]
  const int tid = threadIdx.x, lane = tid & 31, w = tid >> 5;
  const int m0 = blockIdx.x * 128, n0 = blockIdx.y * 64;
  for (int i = tid; i < 2048; i += 256) {             // ctx 128x64 f32 -> f16
    int r = i >> 4, c4 = (i & 15) << 2;
    v4f f = *(const v4f*)(ctx + (size_t)(m0 + r) * 64 + c4);
    v4h hv; hv[0] = (_Float16)f.x; hv[1] = (_Float16)f.y;
            hv[2] = (_Float16)f.z; hv[3] = (_Float16)f.w;
    *(v4h*)(At + r * 80 + c4) = hv;
  }
  for (int i = tid; i < 4096; i += 256) {             // Wfc[k][n] -> Bt[n][k]
    int k = i >> 6, n = i & 63;
    Bt[n * 80 + k] = (_Float16)Wfc[(size_t)k * D_ + n0 + n];
  }
  __syncthreads();
  v8f acc[4] = {};
#pragma unroll
  for (int kt = 0; kt < 2; ++kt) {
    v16h a = load_op_f16(At + w * 16 * 80 + kt * 32, 80, lane);
#pragma unroll
    for (int nt = 0; nt < 4; ++nt) {
      v16h bm = load_op_f16(Bt + nt * 16 * 80 + kt * 32, 80, lane);
      acc[nt] = wmma_f16(a, bm, acc[nt]);
    }
  }
  const int mloc = (lane >> 4) << 3, nloc = lane & 15;
#pragma unroll
  for (int nt = 0; nt < 4; ++nt) {
    int col = n0 + nt * 16 + nloc;
    float bv = bfc[col];
#pragma unroll
    for (int r = 0; r < 8; ++r) {
      int row = m0 + w * 16 + mloc + r;
      out[(size_t)row * D_ + col] = acc[nt][r] + bv + resid[(size_t)row * D_ + col];
    }
  }
}

// ---------------------------------------------------------------------------
// Kernel 5: in-place LayerNorm over rows of D=1024.  One wave per row.
// ---------------------------------------------------------------------------
__global__ __launch_bounds__(256) void k_ln(float* __restrict__ out,
                                            const float* __restrict__ gamma,
                                            const float* __restrict__ beta) {
  const int lane = threadIdx.x & 31, w = threadIdx.x >> 5;
  float* p = out + ((size_t)blockIdx.x * 8 + w) * D_;
  float s = 0.f, ss = 0.f;
  for (int i = lane; i < D_; i += 32) { float x = p[i]; s += x; ss += x * x; }
#pragma unroll
  for (int off = 16; off > 0; off >>= 1) {
    s  += __shfl_xor(s,  off, 32);
    ss += __shfl_xor(ss, off, 32);
  }
  float mu   = s * (1.f / D_);
  float var  = ss * (1.f / D_) - mu * mu;
  float rstd = rsqrtf(var + 1e-5f);
  for (int i = lane; i < D_; i += 32)
    p[i] = (p[i] - mu) * rstd * gamma[i] + beta[i];
}

// ---------------------------------------------------------------------------
extern "C" void kernel_launch(void* const* d_in, const int* in_sizes, int n_in,
                              void* d_out, int out_size, void* d_ws, size_t ws_size,
                              hipStream_t stream) {
  const float* a    = (const float*)d_in[0];
  const float* q    = (const float*)d_in[1];
  const float* k    = (const float*)d_in[2];
  const float* v    = (const float*)d_in[3];
  const unsigned char* mask = (const unsigned char*)d_in[4];
  const float* Wq   = (const float*)d_in[5];
  const float* bq   = (const float*)d_in[6];
  const float* Wk   = (const float*)d_in[7];
  const float* bk   = (const float*)d_in[8];
  const float* Wv   = (const float*)d_in[9];
  const float* bv   = (const float*)d_in[10];
  const float* Wfc  = (const float*)d_in[11];
  const float* bfc  = (const float*)d_in[12];
  const float* gamma= (const float*)d_in[13];
  const float* beta = (const float*)d_in[14];

  float* out  = (float*)d_out;                          // [B, L, D]
  float* attn = (float*)d_out + (size_t)B_ * L_ * D_;   // [H, B, L, L]

  char* ws = (char*)d_ws;
  _Float16* qh  = (_Float16*)(ws);                      // 8 MiB  [B,H,L,64] f16
  _Float16* kh  = (_Float16*)(ws + (size_t)(8  << 20));
  _Float16* vh  = (_Float16*)(ws + (size_t)(16 << 20));
  _Float16* vt  = (_Float16*)(ws + (size_t)(24 << 20)); // 8 MiB  [B,H,64,L] f16
  float*    ctx = (float*)   (ws + (size_t)(32 << 20)); // 1 MiB  [B,L,64] f32

  // allow >default dynamic LDS for the fused attention kernel (CDNA5: 320KB/WGP)
  // (unconditional every call: idempotent, host-side, graph-capture safe)
  (void)hipFuncSetAttribute((const void*)k_attn,
                            hipFuncAttributeMaxDynamicSharedMemorySize,
                            SMEM_ATTN);

  dim3 blk(256);
  k_qkv<<<dim3(32, 16), blk, 0, stream>>>(q, Wq, bq, qh);
  k_qkv<<<dim3(32, 16), blk, 0, stream>>>(k, Wk, bk, kh);
  k_qkv<<<dim3(32, 16), blk, 0, stream>>>(v, Wv, bv, vh);
  k_vtrans<<<dim3(32, 32), blk, 0, stream>>>(vh, vt);

  hipMemsetAsync(ctx, 0, (size_t)B_ * L_ * DV_ * sizeof(float), stream);
  k_attn<<<dim3(128, 32), blk, SMEM_ATTN, stream>>>(qh, kh, vt, mask, a, attn, ctx);

  k_fc<<<dim3(32, 16), blk, 0, stream>>>(ctx, Wfc, bfc, q, out);
  k_ln<<<dim3((B_ * L_) / 8), blk, 0, stream>>>(out, gamma, beta);
}
